// CausalSelfAttention_5368709120804
// MI455X (gfx1250) — matrix operand order repaired and checked by reading, not compiled, not run on hardware
//
#include <hip/hip_runtime.h>

typedef __attribute__((ext_vector_type(8)))  __bf16 bf16x8;
typedef __attribute__((ext_vector_type(16))) __bf16 bf16x16;
typedef __attribute__((ext_vector_type(8)))  float  f32x8;

// ---------------------------------------------------------------------------
// Fragment loaders for CDNA5 WMMA (CDNA5 ISA): contiguous k order, same for A and B.
// A (16x32 bf16, MxK): lane L holds row M=L%16; K = [b..b+15] with
// b = (L<16 ? 0 : 16). Two contiguous 16B loads.
// B (32x16 bf16, KxN): lane L holds column N=L%16; K = [c..c+15] with
// c = (L<16 ? 0 : 16). One contiguous 32B load.
// ---------------------------------------------------------------------------
__device__ __forceinline__ bf16x16 load_a_frag(const __bf16* base, int ldk, int lane) {
    int m    = lane & 15;
    int ksub = (lane < 16) ? 0 : 16;
    const __bf16* p = base + (size_t)m * (size_t)ldk + ksub;
    bf16x8 lo = *(const bf16x8*)(p);
    bf16x8 hi = *(const bf16x8*)(p + 8);
    bf16x16 r;
#pragma unroll
    for (int i = 0; i < 8; ++i) { r[i] = lo[i]; r[i + 8] = hi[i]; }
    return r;
}

__device__ __forceinline__ bf16x16 load_b_frag(const __bf16* base, int ldk, int lane) {
    int n    = lane & 15;
    int koff = (lane < 16) ? 0 : 16;
    const __bf16* p = base + (size_t)n * (size_t)ldk + koff;
    return *(const bf16x16*)(p);
}

__device__ __forceinline__ f32x8 wmma_bf16(bf16x16 a, bf16x16 b, f32x8 c) {
    return __builtin_amdgcn_wmma_f32_16x16x32_bf16(false, a, false, b, (short)0, c,
                                                   false, false);
}

// ---------------------------------------------------------------------------
// Async global -> LDS copy (CDNA5 Tensor/Async path, tracked by ASYNCcnt).
// LDS operand is a byte offset: generic LDS addresses map via addr[31:0]
// (CDNA5 ISA aperture rules). 16B per lane per issue.
// ---------------------------------------------------------------------------
__device__ __forceinline__ void async_copy_b128(const __bf16* src, __bf16* dst_lds) {
    unsigned d = (unsigned)(unsigned long long)dst_lds;
    asm volatile("global_load_async_to_lds_b128 %0, %1, off"
                 :: "v"(d), "v"(src) : "memory");
}
#define ASYNC_WAIT_16() asm volatile("s_wait_asynccnt 0x10" ::: "memory")
#define ASYNC_WAIT_0()  asm volatile("s_wait_asynccnt 0x0"  ::: "memory")

// ---------------------------------------------------------------------------
// fp32 -> bf16 convert
// ---------------------------------------------------------------------------
__global__ void f32_to_bf16_kernel(const float* __restrict__ in,
                                   __bf16* __restrict__ out, int n) {
    int i = blockIdx.x * blockDim.x + threadIdx.x;
    if (i < n) out[i] = (__bf16)in[i];
}

// ---------------------------------------------------------------------------
// C[M,N] = A[M,K] * Bw[N,K]^T + bias[N]  (NT layout; bf16 in, f32 out)
// Block: 256 thr = 8 waves (4x2), block tile 256x128, wave tile 64x64.
// 16 WMMAs per k-step for 8KB of fragment traffic -> 32 FLOP/B from L2.
// ---------------------------------------------------------------------------
__global__ __launch_bounds__(256) void gemm_nt_bias(
    const __bf16* __restrict__ A, const __bf16* __restrict__ Bw,
    const float* __restrict__ bias, float* __restrict__ C,
    int M, int N, int K) {
    int lane = threadIdx.x & 31;
    int wave = threadIdx.x >> 5;
    int wx = wave & 1;
    int wy = wave >> 1;
    int m0 = blockIdx.y * 256 + wy * 64;
    int n0 = blockIdx.x * 128 + wx * 64;

    f32x8 acc[4][4] = {};

    for (int k = 0; k < K; k += 32) {
        bf16x16 a[4], b[4];
#pragma unroll
        for (int i = 0; i < 4; ++i)
            a[i] = load_a_frag(A + (size_t)(m0 + 16 * i) * K + k, K, lane);
#pragma unroll
        for (int i = 0; i < 4; ++i)
            b[i] = load_b_frag(Bw + (size_t)(n0 + 16 * i) * K + k, K, lane);
#pragma unroll
        for (int mi = 0; mi < 4; ++mi)
#pragma unroll
            for (int ni = 0; ni < 4; ++ni)
                acc[mi][ni] = wmma_bf16(a[mi], b[ni], acc[mi][ni]);
    }

    int ncol = lane & 15;
    int half = lane >> 4;
#pragma unroll
    for (int mi = 0; mi < 4; ++mi) {
#pragma unroll
        for (int ni = 0; ni < 4; ++ni) {
            int col = n0 + ni * 16 + ncol;
            float bv = bias[col];
#pragma unroll
            for (int r = 0; r < 8; ++r) {
                int row = m0 + mi * 16 + r + 8 * half;
                C[(size_t)row * N + col] = acc[mi][ni][r] + bv;
            }
        }
    }
}

// ---------------------------------------------------------------------------
// RoPE (reference's exact formula) + head split + bf16 convert.
// qkv layout: [B,T, H*(3*64)]  per head: [q(64) | k(64) | v(64)]
// Outputs: q,k -> [B*H, T, 64] (q pre-scaled by 1/sqrt(64))
//          v   -> transposed [B*H, 64, T]
// ---------------------------------------------------------------------------
__global__ void rope_split_kernel(const float* __restrict__ qkv,
                                  __bf16* __restrict__ qo,
                                  __bf16* __restrict__ ko,
                                  __bf16* __restrict__ vto, int T) {
    int idx = blockIdx.x * blockDim.x + threadIdx.x;
    int p = idx & 31;           // pair index 0..31
    int h = (idx >> 5) & 15;
    int t = (idx >> 9) & 2047;  // T = 2048
    int b = idx >> 20;

    const float* base = qkv + ((size_t)(b * T + t)) * 3072 + h * 192;
    float qe = base[2 * p],       qoo = base[2 * p + 1];
    float ke = base[64 + 2 * p],  koo = base[64 + 2 * p + 1];
    float ve = base[128 + 2 * p], vo  = base[128 + 2 * p + 1];

    float ang = (float)t * __powf(10000.0f, -(float)(2 * p) * (1.0f / 64.0f));
    float s, c;
    __sincosf(ang, &s, &c);
    float q0 = qe * c - qoo * (1.0f - c);
    float q1 = qoo * s + qe * (1.0f - s);
    float k0 = ke * c - koo * (1.0f - c);
    float k1 = koo * s + ke * (1.0f - s);

    const float sm = 0.125f;  // 1/sqrt(DH), folded into q
    size_t bh = (size_t)(b * 16 + h);
    __bf16* qp = qo + (bh * T + t) * 64;
    qp[2 * p]     = (__bf16)(q0 * sm);
    qp[2 * p + 1] = (__bf16)(q1 * sm);
    __bf16* kp = ko + (bh * T + t) * 64;
    kp[2 * p]     = (__bf16)k0;
    kp[2 * p + 1] = (__bf16)k1;
    __bf16* vp = vto + bh * 64 * (size_t)T + t;
    vp[(size_t)(2 * p) * T]     = (__bf16)ve;
    vp[(size_t)(2 * p + 1) * T] = (__bf16)vo;
}

// ---------------------------------------------------------------------------
// Flash attention: one wave per 32-row q tile, s tiles of 32.
// K (32x64) and V^T (64x32) tiles are staged into per-wave LDS with
// double-buffered GLOBAL_LOAD_ASYNC_TO_LDS_B128; the async engine fills the
// next tile while WMMAs consume the current one (s_wait_asynccnt 16 pattern).
// Per s-tile: 8 WMMA (Q*K^T) + 8 WMMA (P*V); P re-layouted through LDS.
// Per-wave LDS: 2x4KB K + 2x4KB V + 2KB P = 18KB; 8 waves = 144KB/block.
// ---------------------------------------------------------------------------
__device__ __forceinline__ void stage_kv_async(const __bf16* kbase,
                                               const __bf16* vtb_s0, int T,
                                               __bf16* kb, __bf16* vb, int lane) {
    // K tile: rows are consecutive -> one contiguous 4KB block
#pragma unroll
    for (int i = 0; i < 8; ++i)
        async_copy_b128(kbase + i * 256 + lane * 8, kb + i * 256 + lane * 8);
    // V^T tile: 64 rows x 32 cols (64B per row), 4 lanes per row
#pragma unroll
    for (int i = 0; i < 8; ++i) {
        int linear = i * 32 + lane;
        int row  = linear >> 2;
        int coff = (linear & 3) * 8;
        async_copy_b128(vtb_s0 + (size_t)row * T + coff, vb + linear * 8);
    }
}

__global__ __launch_bounds__(256) void attn_kernel(
    const __bf16* __restrict__ q, const __bf16* __restrict__ k,
    const __bf16* __restrict__ vt, __bf16* __restrict__ y, int T) {
    __shared__ __attribute__((aligned(32))) __bf16 smem[8][9216];

    int lane = threadIdx.x & 31;
    int wave = threadIdx.x >> 5;
    int bh   = blockIdx.y;                      // 0..31
    int q0   = (blockIdx.x * 8 + wave) * 32;    // 32 q rows per wave
    int ncol = lane & 15;
    int half = lane >> 4;

    __bf16* swave = &smem[wave][0];
    __bf16* kbuf[2] = {swave, swave + 2048};
    __bf16* vbuf[2] = {swave + 4096, swave + 6144};
    __bf16* ptile   = swave + 8192;

    const __bf16* qbase = q + ((size_t)bh * T + q0) * 64;
    bf16x16 qa[2][2];
#pragma unroll
    for (int mt = 0; mt < 2; ++mt) {
        qa[mt][0] = load_a_frag(qbase + (size_t)mt * 16 * 64 + 0,  64, lane);
        qa[mt][1] = load_a_frag(qbase + (size_t)mt * 16 * 64 + 32, 64, lane);
    }

    f32x8 acc[2][4] = {};
    float mrow[2][8], lrow[2][8];
    int rowg[2][8];
#pragma unroll
    for (int mt = 0; mt < 2; ++mt)
#pragma unroll
        for (int r = 0; r < 8; ++r) {
            mrow[mt][r] = -3.0e38f;
            lrow[mt][r] = 0.0f;
            rowg[mt][r] = q0 + mt * 16 + r + 8 * half;
        }

    const __bf16* vtb = vt + (size_t)bh * 64 * T;
    int nst = (q0 >> 5) + 1;  // s tiles of 32 covering [0, q0+32)

    // prologue: stage tile 0 into buffer 0
    stage_kv_async(k + (size_t)bh * T * 64, vtb, T, kbuf[0], vbuf[0], lane);

    for (int st = 0; st < nst; ++st) {
        int cur = st & 1;
        int s0  = st * 32;
        if (st + 1 < nst) {
            // issue next tile, then wait only for the current one (<=16 left)
            stage_kv_async(k + ((size_t)bh * T + s0 + 32) * 64, vtb + s0 + 32, T,
                           kbuf[1 - cur], vbuf[1 - cur], lane);
            ASYNC_WAIT_16();
        } else {
            ASYNC_WAIT_0();
        }

        __bf16* kb = kbuf[cur];
        __bf16* vb = vbuf[cur];

        // K fragments from LDS, shared by both M-tiles
        bf16x16 kb00 = load_b_frag(kb + 0,            64, lane);
        bf16x16 kb01 = load_b_frag(kb + 32,           64, lane);
        bf16x16 kb10 = load_b_frag(kb + 16 * 64 + 0,  64, lane);
        bf16x16 kb11 = load_b_frag(kb + 16 * 64 + 32, 64, lane);

        f32x8 s_lo[2], s_hi[2];
#pragma unroll
        for (int mt = 0; mt < 2; ++mt) {
            f32x8 z = {};
            s_lo[mt] = wmma_bf16(qa[mt][0], kb00, z);
            s_lo[mt] = wmma_bf16(qa[mt][1], kb01, s_lo[mt]);
            s_hi[mt] = wmma_bf16(qa[mt][0], kb10, z);
            s_hi[mt] = wmma_bf16(qa[mt][1], kb11, s_hi[mt]);
        }

        // causal mask + online softmax (scale folded into q)
        int clo = s0 + ncol, chi = clo + 16;
#pragma unroll
        for (int mt = 0; mt < 2; ++mt) {
            float scale_r[8];
#pragma unroll
            for (int r = 0; r < 8; ++r) {
                float vlo = (clo > rowg[mt][r]) ? -3.0e38f : s_lo[mt][r];
                float vhi = (chi > rowg[mt][r]) ? -3.0e38f : s_hi[mt][r];
                float mx = fmaxf(vlo, vhi);
                mx = fmaxf(mx, __shfl_xor(mx, 1, 32));
                mx = fmaxf(mx, __shfl_xor(mx, 2, 32));
                mx = fmaxf(mx, __shfl_xor(mx, 4, 32));
                mx = fmaxf(mx, __shfl_xor(mx, 8, 32));
                float mnew = fmaxf(mrow[mt][r], mx);
                float sc   = __expf(mrow[mt][r] - mnew);
                float plo  = __expf(vlo - mnew);
                float phi  = __expf(vhi - mnew);
                float rs = plo + phi;
                rs += __shfl_xor(rs, 1, 32);
                rs += __shfl_xor(rs, 2, 32);
                rs += __shfl_xor(rs, 4, 32);
                rs += __shfl_xor(rs, 8, 32);
                lrow[mt][r] = lrow[mt][r] * sc + rs;
                mrow[mt][r] = mnew;
                scale_r[r]  = sc;
                s_lo[mt][r] = plo;
                s_hi[mt][r] = phi;
            }
#pragma unroll
            for (int c = 0; c < 4; ++c)
#pragma unroll
                for (int r = 0; r < 8; ++r) acc[mt][c][r] *= scale_r[r];

            // stage P (16x32 bf16, row-major) in per-wave LDS
#pragma unroll
            for (int r = 0; r < 8; ++r) {
                int row = mt * 16 + r + 8 * half;
                ptile[row * 32 + ncol]      = (__bf16)s_lo[mt][r];
                ptile[row * 32 + ncol + 16] = (__bf16)s_hi[mt][r];
            }
        }

        // P A-fragments from LDS; V fragments from LDS, shared by both M-tiles
        bf16x16 pa0 = load_a_frag(ptile + 0,       32, lane);
        bf16x16 pa1 = load_a_frag(ptile + 16 * 32, 32, lane);
#pragma unroll
        for (int c = 0; c < 4; ++c) {
            bf16x16 vfrag = load_b_frag(vb + 16 * c * 32, 32, lane);
            acc[0][c] = wmma_bf16(pa0, vfrag, acc[0][c]);
            acc[1][c] = wmma_bf16(pa1, vfrag, acc[1][c]);
        }
    }

    // epilogue: y[b, q0+row, h*64 + dh] bf16
    int b = bh >> 4, h = bh & 15;
#pragma unroll
    for (int mt = 0; mt < 2; ++mt) {
        __bf16* yb = y + (size_t)(b * T + q0 + mt * 16) * 1024 + h * 64 + ncol;
#pragma unroll
        for (int r = 0; r < 8; ++r) {
            int row = r + 8 * half;
            float inv = 1.0f / lrow[mt][r];
            __bf16* pr = yb + (size_t)row * 1024;
            pr[0]  = (__bf16)(acc[mt][0][r] * inv);
            pr[16] = (__bf16)(acc[mt][1][r] * inv);
            pr[32] = (__bf16)(acc[mt][2][r] * inv);
            pr[48] = (__bf16)(acc[mt][3][r] * inv);
        }
    }
}

// ---------------------------------------------------------------------------
// Launcher
// ---------------------------------------------------------------------------
extern "C" void kernel_launch(void* const* d_in, const int* in_sizes, int n_in,
                              void* d_out, int out_size, void* d_ws, size_t ws_size,
                              hipStream_t stream) {
    const float* x     = (const float*)d_in[0];
    const float* Wqkv  = (const float*)d_in[1];
    const float* bqkv  = (const float*)d_in[2];
    const float* Wproj = (const float*)d_in[3];
    const float* bproj = (const float*)d_in[4];
    float* out = (float*)d_out;

    const int Bc = 2, Tc = 2048, Dc = 1024, Hc = 16;
    const int Mrows = Bc * Tc;  // 4096

    char* ws = (char*)d_ws;
    const size_t MB = 1ull << 20;
    __bf16* x_bf     = (__bf16*)(ws + 0 * MB);   //  8 MB
    __bf16* wqkv_bf  = (__bf16*)(ws + 8 * MB);   //  6 MB
    __bf16* wproj_bf = (__bf16*)(ws + 14 * MB);  //  2 MB
    float*  qkv      = (float*)(ws + 16 * MB);   // 48 MB
    __bf16* q_bf     = (__bf16*)(ws + 64 * MB);  //  8 MB
    __bf16* k_bf     = (__bf16*)(ws + 72 * MB);  //  8 MB
    __bf16* vt_bf    = (__bf16*)(ws + 80 * MB);  //  8 MB
    __bf16* y_bf     = (__bf16*)(ws + 88 * MB);  //  8 MB

    int nx = Mrows * Dc;
    f32_to_bf16_kernel<<<(nx + 255) / 256, 256, 0, stream>>>(x, x_bf, nx);
    int nw1 = 3 * Dc * Dc;
    f32_to_bf16_kernel<<<(nw1 + 255) / 256, 256, 0, stream>>>(Wqkv, wqkv_bf, nw1);
    int nw2 = Dc * Dc;
    f32_to_bf16_kernel<<<(nw2 + 255) / 256, 256, 0, stream>>>(Wproj, wproj_bf, nw2);

    // qkv = x @ Wqkv^T + bqkv : M=4096, N=3072, K=1024
    dim3 g1((3 * Dc) / 128, Mrows / 256);
    gemm_nt_bias<<<g1, 256, 0, stream>>>(x_bf, wqkv_bf, bqkv, qkv,
                                         Mrows, 3 * Dc, Dc);

    // RoPE + split + transpose V
    int nr = Bc * Tc * Hc * 32;
    rope_split_kernel<<<nr / 256, 256, 0, stream>>>(qkv, q_bf, k_bf, vt_bf, Tc);

    // flash attention: 64 q-tiles (32 rows) per (b,h), 8 waves per block
    dim3 g2((Tc / 32) / 8, Bc * Hc);
    attn_kernel<<<g2, 256, 0, stream>>>(q_bf, k_bf, vt_bf, y_bf, Tc);

    // out = y @ Wproj^T + bproj : M=4096, N=1024, K=1024
    dim3 g3(Dc / 128, Mrows / 256);
    gemm_nt_bias<<<g3, 256, 0, stream>>>(y_bf, wproj_bf, bproj, out,
                                         Mrows, Dc, Dc);
}
